// Mamba2DecoderLayer_7301444403551
// MI455X (gfx1250) — matrix operand order repaired and checked by reading, not compiled, not run on hardware
//
#include <hip/hip_runtime.h>
#include <hip/hip_bf16.h>
#include <math.h>

#define BATCH   2
#define SEQ     2048
#define HIDDEN  2048
#define NHEADS  64
#define HEADDIM 64
#define DSTATE  128
#define DCONV   4
#define DINNER  (NHEADS*HEADDIM)                 // 4096
#define CONVDIM (DINNER + 2*DSTATE)              // 4352
#define DINPROJ (2*DINNER + 2*DSTATE + NHEADS)   // 8512
#define NTOK    (BATCH*SEQ)                      // 4096
#define EPS     1e-5f

typedef __attribute__((ext_vector_type(16))) __bf16 v16bf;
typedef __attribute__((ext_vector_type(8)))  float  v8f;

// ---------------- helpers ----------------

__device__ __forceinline__ float warp_sum(float v) {
  #pragma unroll
  for (int off = 16; off > 0; off >>= 1) v += __shfl_xor(v, off, 32);
  return v;
}

__device__ __forceinline__ float block_sum_256(float v, float* sh) {
  v = warp_sum(v);
  const int lane = threadIdx.x & 31;
  const int w    = threadIdx.x >> 5;
  if (lane == 0) sh[w] = v;
  __syncthreads();
  float t = (threadIdx.x < 8) ? sh[threadIdx.x] : 0.0f;
  if (w == 0) {
    t = warp_sum(t);
    if (lane == 0) sh[0] = t;
  }
  __syncthreads();
  float r = sh[0];
  __syncthreads();
  return r;
}

__device__ __forceinline__ float silu_f(float x) {
  return x / (1.0f + __expf(-x));
}

// ---------------- kernel 1: x = hs + res ; new_residual = x ; h_bf = rmsnorm(x)*w ----------------
__global__ void __launch_bounds__(256)
k_add_rmsnorm(const float* __restrict__ hs, const float* __restrict__ res,
              const float* __restrict__ w, float* __restrict__ resid_out,
              __bf16* __restrict__ hbf) {
  __shared__ float sh[8];
  const int tok = blockIdx.x;
  const float* a = hs  + (size_t)tok * HIDDEN;
  const float* b = res + (size_t)tok * HIDDEN;
  float x[8];
  float ss = 0.0f;
  #pragma unroll
  for (int i = 0; i < 8; i++) {
    const int c = threadIdx.x + i * 256;
    x[i] = a[c] + b[c];
    resid_out[(size_t)tok * HIDDEN + c] = x[i];
    ss += x[i] * x[i];
  }
  ss = block_sum_256(ss, sh);
  const float scale = rsqrtf(ss / (float)HIDDEN + EPS);
  #pragma unroll
  for (int i = 0; i < 8; i++) {
    const int c = threadIdx.x + i * 256;
    hbf[(size_t)tok * HIDDEN + c] = (__bf16)(x[i] * scale * w[c]);
  }
}

// ---------------- kernel 2: f32 -> bf16 convert ----------------
__global__ void __launch_bounds__(256)
k_f32_to_bf16(const float* __restrict__ in, __bf16* __restrict__ out, int n) {
  const int i = blockIdx.x * 256 + threadIdx.x;
  if (i < n) out[i] = (__bf16)in[i];
}

// ---------------- kernel 3: bf16 WMMA GEMM: C[M,N] = A[M,K] * W[N,K]^T ----------------
// wave tile 32(M)x64(N) (2 A frags x 4 B frags = 8 WMMA/K-step).
// block = 8 waves stacked in M -> 256x64 block tile.
// B tile (64N x 32K bf16 = 4KB) staged in LDS via async-to-LDS, double buffered;
// steady-state loop uses constant s_wait_asynccnt 0x1, final iteration peeled.
// requires M%256==0, N%64==0, K%64==0 (true for all calls here).
#define BPAD 40  // padded row length (elements): 80B = 20 banks, breaks conflicts, 16B aligned

__global__ void __launch_bounds__(256)
k_gemm_bf16_wmma(const __bf16* __restrict__ A, const __bf16* __restrict__ W,
                 float* __restrict__ C, int M, int N, int K) {
  __shared__ __bf16 Bsh[2][64][BPAD];

  const int tid  = threadIdx.x;
  const int lane = tid & 31;
  const int wave = tid >> 5;
  const int m0 = blockIdx.y * 256 + wave * 32;
  const int n0 = blockIdx.x * 64;
  const int lm = lane & 15;           // row (A) / col (B) within 16-wide tile
  const int kh = (lane >> 4) * 16;    // K half: lanes 0-15 -> K 0..15, lanes 16-31 -> 16..31

  // staging: thread t loads B row n = t>>2, k-part (t&3)*8  (16B = one b128)
  const int sn = tid >> 2;
  const int sk = (tid & 3) * 8;
  const __bf16* Wst = W + (size_t)(n0 + sn) * K + sk;
  const unsigned lds_base0 = (unsigned)(size_t)&Bsh[0][sn][sk];
  const unsigned lds_base1 = (unsigned)(size_t)&Bsh[1][sn][sk];

  v8f acc[2][4] = {};

  const __bf16* Arow0 = A + (size_t)(m0 + lm)      * K + kh;
  const __bf16* Arow1 = A + (size_t)(m0 + 16 + lm) * K + kh;

  // prologue: stage k=0 into buf0, k=32 into buf1
  asm volatile("global_load_async_to_lds_b128 %0, %1, off"
               :: "v"(lds_base0), "v"((const void*)(Wst)) : "memory");
  asm volatile("global_load_async_to_lds_b128 %0, %1, off"
               :: "v"(lds_base1), "v"((const void*)(Wst + 32)) : "memory");

  v16bf a0 = *(const v16bf*)(Arow0);
  v16bf a1 = *(const v16bf*)(Arow1);

  // steady state: k in [0, K-32), next tile always in flight
  int k = 0;
  for (; k + 32 < K; k += 32) {
    __builtin_prefetch(Arow0 + k + 1024, 0, 1);
    v16bf a0n = *(const v16bf*)(Arow0 + k + 32);
    v16bf a1n = *(const v16bf*)(Arow1 + k + 32);

    asm volatile("s_wait_asynccnt 0x1" ::: "memory");  // oldest tile (cur buf) arrived
    __syncthreads();

    const int cur = (k >> 5) & 1;
    v16bf b[4];
    #pragma unroll
    for (int j = 0; j < 4; j++) {
      b[j] = *(const v16bf*)&Bsh[cur][j * 16 + lm][kh];
    }
    #pragma unroll
    for (int j = 0; j < 4; j++) {
      acc[0][j] = __builtin_amdgcn_wmma_f32_16x16x32_bf16(
          false, a0, false, b[j], (short)0, acc[0][j], false, false);
      acc[1][j] = __builtin_amdgcn_wmma_f32_16x16x32_bf16(
          false, a1, false, b[j], (short)0, acc[1][j], false, false);
    }
    __syncthreads();  // all waves done reading buf[cur] before overwrite

    // branchless: stage k+64 (or a harmless redundant tile on the last steady iter;
    // it is drained by the final s_wait_asynccnt 0x0 and never read)
    const int knext = (k + 64 < K) ? (k + 64) : 0;
    const unsigned lds = cur ? lds_base1 : lds_base0;
    asm volatile("global_load_async_to_lds_b128 %0, %1, off"
                 :: "v"(lds), "v"((const void*)(Wst + knext)) : "memory");
    a0 = a0n;
    a1 = a1n;
  }

  // peeled final iteration (k == K-32)
  asm volatile("s_wait_asynccnt 0x0" ::: "memory");
  __syncthreads();
  {
    const int cur = (k >> 5) & 1;
    v16bf b[4];
    #pragma unroll
    for (int j = 0; j < 4; j++) {
      b[j] = *(const v16bf*)&Bsh[cur][j * 16 + lm][kh];
    }
    #pragma unroll
    for (int j = 0; j < 4; j++) {
      acc[0][j] = __builtin_amdgcn_wmma_f32_16x16x32_bf16(
          false, a0, false, b[j], (short)0, acc[0][j], false, false);
      acc[1][j] = __builtin_amdgcn_wmma_f32_16x16x32_bf16(
          false, a1, false, b[j], (short)0, acc[1][j], false, false);
    }
  }

  // D layout: VGPR r -> row (lane>>4)*8 + r, col (lane&15)
  const int rsub = (lane >> 4) * 8;
  #pragma unroll
  for (int ai = 0; ai < 2; ai++) {
    #pragma unroll
    for (int j = 0; j < 4; j++) {
      float* Cp = C + (size_t)(m0 + ai * 16 + rsub) * N + n0 + j * 16 + lm;
      #pragma unroll
      for (int r = 0; r < 8; r++) Cp[(size_t)r * N] = acc[ai][j][r];
    }
  }
}

// ---------------- kernel 4: dt = softplus(dt_raw + bias); decay = exp(-exp(A_log)*dt) ----------------
__global__ void __launch_bounds__(256)
k_dt(const float* __restrict__ zx, const float* __restrict__ dt_bias,
     const float* __restrict__ A_log, float* __restrict__ dtp,
     float* __restrict__ decay) {
  const int i = blockIdx.x * 256 + threadIdx.x;
  if (i >= NTOK * NHEADS) return;
  const int t = i / NHEADS;
  const int h = i - t * NHEADS;
  const float raw = zx[(size_t)t * DINPROJ + (DINNER + CONVDIM) + h] + dt_bias[h];
  const float sp  = (raw > 20.0f) ? raw : log1pf(__expf(raw));
  dtp[i]   = sp;
  decay[i] = __expf(-__expf(A_log[h]) * sp);
}

// ---------------- kernel 5: causal depthwise conv (DCONV=4) + SiLU ----------------
__global__ void __launch_bounds__(256)
k_conv(const float* __restrict__ zx, const float* __restrict__ cw,
       const float* __restrict__ cb, float* __restrict__ xbc) {
  const int i = blockIdx.x * 256 + threadIdx.x;
  if (i >= NTOK * CONVDIM) return;
  const int c   = i % CONVDIM;
  const int tok = i / CONVDIM;
  const int b   = tok / SEQ;
  const int s   = tok - b * SEQ;
  float acc = cb[c];
  #pragma unroll
  for (int k = 0; k < DCONV; k++) {
    const int ss = s - (DCONV - 1) + k;
    if (ss >= 0) {
      acc += zx[((size_t)(b * SEQ + ss)) * DINPROJ + DINNER + c] * cw[c * DCONV + k];
    }
  }
  xbc[(size_t)tok * CONVDIM + c] = silu_f(acc);
}

// ---------------- kernel 6: SSM scan ----------------
// one block per (batch, head); state 64x128 kept in registers (32 f32/thread).
__global__ void __launch_bounds__(256)
k_scan(const float* __restrict__ xbc, const float* __restrict__ dtp,
       const float* __restrict__ decay, const float* __restrict__ Dp,
       float* __restrict__ y) {
  __shared__ float xs_sh[HEADDIM];
  __shared__ float dB_sh[DSTATE];
  __shared__ float C_sh[DSTATE];

  const int h = blockIdx.x & (NHEADS - 1);
  const int b = blockIdx.x >> 6;
  const int tid = threadIdx.x;
  const int p = tid >> 2;
  const int q = tid & 3;
  const float Dh = Dp[h];

  float st[32];
  #pragma unroll
  for (int i = 0; i < 32; i++) st[i] = 0.0f;

  for (int s = 0; s < SEQ; s++) {
    const size_t tok  = (size_t)b * SEQ + s;
    const size_t base = tok * CONVDIM;
    const float dt_s  = dtp[tok * NHEADS + h];
    const float dec   = decay[tok * NHEADS + h];

    if (tid < HEADDIM) {
      xs_sh[tid] = xbc[base + (size_t)h * HEADDIM + tid];
    } else if (tid < HEADDIM + DSTATE) {
      const int n = tid - HEADDIM;
      dB_sh[n] = dt_s * xbc[base + DINNER + n];
    } else {
      const int n = (tid - (HEADDIM + DSTATE)) * 2;
      C_sh[n]     = xbc[base + DINNER + DSTATE + n];
      C_sh[n + 1] = xbc[base + DINNER + DSTATE + n + 1];
    }
    __syncthreads();

    const float xp = xs_sh[p];
    float acc = 0.0f;
    const int nb = q * 32;
    #pragma unroll
    for (int i = 0; i < 32; i++) {
      st[i] = st[i] * dec + xp * dB_sh[nb + i];
      acc += st[i] * C_sh[nb + i];
    }
    acc += __shfl_xor(acc, 1, 32);
    acc += __shfl_xor(acc, 2, 32);
    if (q == 0) {
      y[tok * DINNER + (size_t)h * HEADDIM + p] = acc + Dh * xp;
    }
    __syncthreads();
  }
}

// ---------------- kernel 7: gate + rmsnorm -> bf16 ----------------
__global__ void __launch_bounds__(256)
k_gate_rmsnorm(const float* __restrict__ y, const float* __restrict__ zx,
               const float* __restrict__ gw, __bf16* __restrict__ ybf) {
  __shared__ float sh[8];
  const int tok = blockIdx.x;
  float g[16];
  float ss = 0.0f;
  #pragma unroll
  for (int i = 0; i < 16; i++) {
    const int c = threadIdx.x + i * 256;
    const float z = zx[(size_t)tok * DINPROJ + c];
    g[i] = y[(size_t)tok * DINNER + c] * silu_f(z);
    ss += g[i] * g[i];
  }
  ss = block_sum_256(ss, sh);
  const float scale = rsqrtf(ss / (float)DINNER + EPS);
  #pragma unroll
  for (int i = 0; i < 16; i++) {
    const int c = threadIdx.x + i * 256;
    ybf[(size_t)tok * DINNER + c] = (__bf16)(g[i] * scale * gw[c]);
  }
}

// ---------------- host launcher ----------------
extern "C" void kernel_launch(void* const* d_in, const int* in_sizes, int n_in,
                              void* d_out, int out_size, void* d_ws, size_t ws_size,
                              hipStream_t stream) {
  const float* hs        = (const float*)d_in[0];
  const float* residual  = (const float*)d_in[1];
  const float* norm_w    = (const float*)d_in[2];
  const float* in_proj_w = (const float*)d_in[3];
  const float* conv_w    = (const float*)d_in[4];
  const float* conv_b    = (const float*)d_in[5];
  const float* A_log     = (const float*)d_in[6];
  const float* dt_bias   = (const float*)d_in[7];
  const float* Dp        = (const float*)d_in[8];
  const float* gate_w    = (const float*)d_in[9];
  const float* out_proj_w= (const float*)d_in[10];

  float* out_main = (float*)d_out;                              // [NTOK, HIDDEN]
  float* out_res  = (float*)d_out + (size_t)NTOK * HIDDEN;      // [NTOK, HIDDEN]

  char* ws = (char*)d_ws;
  size_t off = 0;
  auto alloc = [&](size_t bytes) { size_t o = off; off += (bytes + 255) & ~(size_t)255; return o; };
  const size_t off_hbf  = alloc((size_t)NTOK * HIDDEN * 2);            // h bf16
  const size_t off_w1   = alloc((size_t)DINPROJ * HIDDEN * 2);         // in_proj bf16
  const size_t off_w2   = alloc((size_t)HIDDEN * DINNER * 2);          // out_proj bf16
  const size_t off_zx   = alloc((size_t)NTOK * DINPROJ * 4);           // zxbcdt f32
  const size_t off_xbc  = alloc((size_t)NTOK * CONVDIM * 4);           // conv+silu f32
  const size_t off_dtp  = alloc((size_t)NTOK * NHEADS * 4);
  const size_t off_dec  = alloc((size_t)NTOK * NHEADS * 4);
  const size_t off_y    = alloc((size_t)NTOK * DINNER * 4);            // scan output f32
  const size_t off_ybf  = alloc((size_t)NTOK * DINNER * 2);            // gated bf16
  (void)ws_size;

  __bf16* hbf  = (__bf16*)(ws + off_hbf);
  __bf16* w1bf = (__bf16*)(ws + off_w1);
  __bf16* w2bf = (__bf16*)(ws + off_w2);
  float*  zx   = (float*)(ws + off_zx);
  float*  xbc  = (float*)(ws + off_xbc);
  float*  dtp  = (float*)(ws + off_dtp);
  float*  dec  = (float*)(ws + off_dec);
  float*  ybuf = (float*)(ws + off_y);
  __bf16* ybf  = (__bf16*)(ws + off_ybf);

  // 1) residual add + rmsnorm -> hbf, out_res
  k_add_rmsnorm<<<NTOK, 256, 0, stream>>>(hs, residual, norm_w, out_res, hbf);

  // 2) weight conversions to bf16
  {
    const int n1 = DINPROJ * HIDDEN;
    k_f32_to_bf16<<<(n1 + 255) / 256, 256, 0, stream>>>(in_proj_w, w1bf, n1);
    const int n2 = HIDDEN * DINNER;
    k_f32_to_bf16<<<(n2 + 255) / 256, 256, 0, stream>>>(out_proj_w, w2bf, n2);
  }

  // 3) GEMM1: zx[NTOK, DINPROJ] = hbf[NTOK, HIDDEN] * w1bf[DINPROJ, HIDDEN]^T
  {
    dim3 grid(DINPROJ / 64, NTOK / 256);
    k_gemm_bf16_wmma<<<grid, 256, 0, stream>>>(hbf, w1bf, zx, NTOK, DINPROJ, HIDDEN);
  }

  // 4) dt / decay precompute
  {
    const int n = NTOK * NHEADS;
    k_dt<<<(n + 255) / 256, 256, 0, stream>>>(zx, dt_bias, A_log, dtp, dec);
  }

  // 5) causal depthwise conv + silu
  {
    const int n = NTOK * CONVDIM;
    k_conv<<<(n + 255) / 256, 256, 0, stream>>>(zx, conv_w, conv_b, xbc);
  }

  // 6) SSM scan (128 blocks = BATCH*NHEADS)
  k_scan<<<BATCH * NHEADS, 256, 0, stream>>>(xbc, dtp, dec, Dp, ybuf);

  // 7) gate + rmsnorm -> bf16
  k_gate_rmsnorm<<<NTOK, 256, 0, stream>>>(ybuf, zx, gate_w, ybf);

  // 8) GEMM2: out[NTOK, HIDDEN] = ybf[NTOK, DINNER] * w2bf[HIDDEN, DINNER]^T
  {
    dim3 grid(HIDDEN / 64, NTOK / 256);
    k_gemm_bf16_wmma<<<grid, 256, 0, stream>>>(ybf, w2bf, out_main, NTOK, HIDDEN, DINNER);
  }
}
